// attentionblock_39848706573411
// MI455X (gfx1250) — matrix operand order repaired and checked by reading, not compiled, not run on hardware
//
#include <hip/hip_runtime.h>
#include <math.h>

// ---------------------------------------------------------------------------
// AttentionBlock: q=conv3x3(x,wq) [B,16,N], k=conv3x3(y,wk) [B,16,N],
// v=conv3x3(y,wv) [B,256,N]; out[b,c,i] = sum_j v[b,c,j]*softmax_j(q[:,i].k[:,j])
// B=4, C=256, H=W=64, N=4096, CQK=16.
// Strategy: bf16 WMMA implicit-GEMM convs + fused flash attention (bf16 WMMA,
// f32 accum), LDS-staged V tiles via gfx1250 async global->LDS (guarded).
// ---------------------------------------------------------------------------

typedef __attribute__((ext_vector_type(16))) __bf16 v16bf;
typedef __attribute__((ext_vector_type(8)))  __bf16 v8bf;
typedef __attribute__((ext_vector_type(8)))  float  v8f;
typedef __attribute__((ext_vector_type(4)))  int    v4i;

union V16U { v16bf v; v8bf h[2]; __bf16 e[16]; };
union V8BU { v8bf v; __bf16 e[8]; };

#define NB   4
#define CCH  256
#define HH_  64
#define WW_  64
#define NTOK 4096
#define DQK  16
#define BJ   32     // j-tile per attention step

// ---------------- async global -> LDS (CDNA5) with compile-safe fallback ----
#if defined(__AMDGCN__) && __has_builtin(__builtin_amdgcn_global_load_async_to_lds_b128) && __has_builtin(__builtin_amdgcn_s_wait_asynccnt)
#define USE_ASYNC_LDS 1
#else
#define USE_ASYNC_LDS 0
#endif

__device__ __forceinline__ void cp_async16(__bf16* lds_dst, const __bf16* g_src) {
#if USE_ASYNC_LDS
  // toolchain prototype: (v4i* global_src, v4i* lds_dst, imm offset, imm cpol)
  __builtin_amdgcn_global_load_async_to_lds_b128((v4i*)g_src, (v4i*)lds_dst, 0, 0);
#else
  *(uint4*)lds_dst = *(const uint4*)g_src;
#endif
}

#if USE_ASYNC_LDS
#define ASYNC_WAIT(n) __builtin_amdgcn_s_wait_asynccnt(n)
#else
#define ASYNC_WAIT(n) ((void)0)
#endif

__device__ __forceinline__ void lds_fence() {
#if defined(__AMDGCN__)
  asm volatile("s_wait_dscnt 0" ::: "memory");
#endif
}

// ---------------- packing kernels ------------------------------------------
// NCHW f32 -> NHWC bf16 (exactly 4*64*64*256 = 4194304 elements)
__global__ void pack_nhwc_kernel(const float* __restrict__ in, __bf16* __restrict__ out) {
  int i = blockIdx.x * 256 + threadIdx.x;          // output-linear [b][h][w][c]
  int c = i & 255; int t = i >> 8;
  int w = t & 63;  t >>= 6;
  int h = t & 63;  int b = t >> 6;
  out[i] = (__bf16) in[(((size_t)b * CCH + c) * HH_ + h) * WW_ + w];
}

// w [OC][256][3][3] f32 -> wpack [9][OC][256] bf16
__global__ void pack_w_kernel(const float* __restrict__ w, __bf16* __restrict__ wp, int OC) {
  int i = blockIdx.x * 256 + threadIdx.x;
  int total = OC * CCH * 9;
  if (i >= total) return;
  int ci = i & 255; int t = i >> 8;
  int oc = t % OC;  int tap = t / OC;
  wp[i] = (__bf16) w[((size_t)(oc * CCH + ci)) * 9 + tap];
}

// ---------------- implicit-GEMM conv3x3 (SAME), bf16 WMMA ------------------
// in: NHWC bf16. wpack: [9][OC][256] bf16. Each wave: 16 out-channels x 16 pixels.
// mode 0: out[b][c][n] (channel-major, used for V^T), mode 1: out[b][n][oc] (OC=16)
__global__ __launch_bounds__(256) void conv_wmma_kernel(
    const __bf16* __restrict__ in, const __bf16* __restrict__ wpack,
    const float* __restrict__ bias, __bf16* __restrict__ outp,
    int OC, int mode) {
  const int cTiles = OC >> 4;
  int tile = blockIdx.x * 8 + (threadIdx.x >> 5);
  const int lane = threadIdx.x & 31;
  const int l16  = lane & 15;
  const bool hiL = lane >= 16;

  const int ct = tile % cTiles; tile /= cTiles;
  const int pt = tile & 3;      tile >>= 2;
  const int h  = tile & 63;
  const int b  = tile >> 6;
  const int c0 = ct * 16;
  const int w0 = pt * 16;

  v8f acc;
  #pragma unroll
  for (int r = 0; r < 8; ++r) acc[r] = 0.f;

  #pragma unroll
  for (int kh = 0; kh < 3; ++kh) {
    const int hh = h + kh - 1;
    const bool okh = (hh >= 0) && (hh < HH_);
    #pragma unroll
    for (int kw = 0; kw < 3; ++kw) {
      const int ww = w0 + l16 + kw - 1;
      const bool ok = okh && (ww >= 0) && (ww < WW_);
      const int tap = kh * 3 + kw;
      // A rows (weights): lane half selects k range 0..15 / 16..31 within 32-chunk
      const __bf16* wrow = wpack + ((size_t)tap * OC + c0 + l16) * CCH + (hiL ? 16 : 0);
      // B cols (pixels): lane half selects k range 0..15 / 16..31
      const __bf16* brow = in + (((size_t)(b * HH_ + hh) * WW_ + ww) * CCH) + (hiL ? 16 : 0);
      #pragma unroll
      for (int cb = 0; cb < 8; ++cb) {          // 32 input channels per step
        V16U a, bt;
        a.h[0] = *(const v8bf*)(wrow + cb * 32);
        a.h[1] = *(const v8bf*)(wrow + cb * 32 + 8);
        #pragma unroll
        for (int q = 0; q < 16; ++q) bt.e[q] = (__bf16)0.f;
        if (ok) bt.v = *(const v16bf*)(brow + cb * 32);
        acc = __builtin_amdgcn_wmma_f32_16x16x32_bf16(
            false, a.v, false, bt.v, (short)0, acc, false, false);
      }
    }
  }

  const int pix = l16;
  if (mode == 0) {
    const size_t n = (size_t)h * WW_ + w0 + pix;
    #pragma unroll
    for (int r = 0; r < 8; ++r) {
      const int cl = c0 + r + (hiL ? 8 : 0);
      outp[((size_t)b * OC + cl) * NTOK + n] = (__bf16)(acc[r] + bias[cl]);
    }
  } else {
    V8BU pk;
    #pragma unroll
    for (int r = 0; r < 8; ++r) {
      const int cl = c0 + r + (hiL ? 8 : 0);
      pk.e[r] = (__bf16)(acc[r] + bias[cl]);
    }
    *(v8bf*)(outp + (((size_t)b * NTOK + (size_t)h * WW_ + w0 + pix) * DQK) + (hiL ? 8 : 0)) = pk.v;
  }
}

// ---------------- fused flash attention ------------------------------------
// Qp,Kp: [B][N][16] bf16 (position-major). Vt: [B][C][N] bf16 (v transposed in j).
// Block = 128 threads (4 waves); wave handles a 16-row i-tile; block shares
// double-buffered LDS V tiles [256][32], filled with async global->LDS.
__global__ __launch_bounds__(128, 1) void attn_kernel(
    const __bf16* __restrict__ Qp, const __bf16* __restrict__ Kp,
    const __bf16* __restrict__ Vt, float* __restrict__ out) {
  __shared__ __align__(64) __bf16 Vbuf[2][CCH * BJ];   // 2 x 16 KB
  __shared__ __align__(64) __bf16 Pbuf[4][16 * BJ];    // 4 x 1 KB (wave-private)

  const int tid  = threadIdx.x;
  const int wave = tid >> 5;
  const int lane = tid & 31;
  const int l16  = lane & 15;
  const bool hiL = lane >= 16;

  const int b  = blockIdx.x >> 6;                 // 64 blocks per batch
  const int i0 = (blockIdx.x & 63) * 64 + wave * 16;

  // Q tile in A-layout, K-dim padded 16 -> 32 with zeros
  V16U aQ;
  {
    const __bf16* qrow = Qp + ((size_t)b * NTOK + i0 + l16) * DQK;
    #pragma unroll
    for (int q = 0; q < 16; ++q) aQ.e[q] = (__bf16)0.f;
    if (!hiL) aQ.v = *(const v16bf*)qrow;
  }

  v8f acc[16];
  #pragma unroll
  for (int c = 0; c < 16; ++c)
    #pragma unroll
    for (int r = 0; r < 8; ++r) acc[c][r] = 0.f;
  float mrow[8], lrow[8];
  #pragma unroll
  for (int r = 0; r < 8; ++r) { mrow[r] = -__builtin_inff(); lrow[r] = 0.f; }

  auto copyTile = [&](int buf, int j0) {
    // 256 rows x 64B = 16 KB; 1024 x 16B segments over 128 threads => 8 each
    #pragma unroll
    for (int q = 0; q < 8; ++q) {
      const int idx = tid * 8 + q;
      const int c = idx >> 2, seg = idx & 3;
      cp_async16(&Vbuf[buf][c * BJ + seg * 8],
                 Vt + ((size_t)b * CCH + c) * NTOK + j0 + seg * 8);
    }
  };

  const int NSTEP = NTOK / BJ;   // 128
  copyTile(0, 0);

  for (int t = 0; t < NSTEP; ++t) {
    const int j0 = t * BJ;
    const bool more = (t + 1) < NSTEP;
    if (more) copyTile((t + 1) & 1, j0 + BJ);
    if (more) ASYNC_WAIT(8); else ASYNC_WAIT(0);
    __syncthreads();
    const __bf16* vb = &Vbuf[t & 1][0];

    // K^T tiles in B-layout: lanes 0-15 hold k=0..15 (all 16 d's), hi lanes zero
    V16U bK0, bK1;
    #pragma unroll
    for (int q = 0; q < 16; ++q) { bK0.e[q] = (__bf16)0.f; bK1.e[q] = (__bf16)0.f; }
    if (!hiL) {
      bK0.v = *(const v16bf*)(Kp + ((size_t)b * NTOK + j0 + l16) * DQK);
      bK1.v = *(const v16bf*)(Kp + ((size_t)b * NTOK + j0 + 16 + l16) * DQK);
    }
    if (more) __builtin_prefetch(Kp + ((size_t)b * NTOK + j0 + BJ) * DQK, 0, 1);

    v8f zc;
    #pragma unroll
    for (int r = 0; r < 8; ++r) zc[r] = 0.f;
    v8f S0 = __builtin_amdgcn_wmma_f32_16x16x32_bf16(false, aQ.v, false, bK0.v, (short)0, zc, false, false);
    v8f S1 = __builtin_amdgcn_wmma_f32_16x16x32_bf16(false, aQ.v, false, bK1.v, (short)0, zc, false, false);

    // online softmax; row reductions stay within each 16-lane half (xor 1,2,4,8)
    float alpha[8], P0[8], P1[8];
    #pragma unroll
    for (int r = 0; r < 8; ++r) {
      float mx = fmaxf(S0[r], S1[r]);
      mx = fmaxf(mx, __shfl_xor(mx, 1, 32));
      mx = fmaxf(mx, __shfl_xor(mx, 2, 32));
      mx = fmaxf(mx, __shfl_xor(mx, 4, 32));
      mx = fmaxf(mx, __shfl_xor(mx, 8, 32));
      const float nm = fmaxf(mrow[r], mx);
      alpha[r] = __expf(mrow[r] - nm);
      mrow[r]  = nm;
      P0[r] = __expf(S0[r] - nm);
      P1[r] = __expf(S1[r] - nm);
      float s = P0[r] + P1[r];
      s += __shfl_xor(s, 1, 32);
      s += __shfl_xor(s, 2, 32);
      s += __shfl_xor(s, 4, 32);
      s += __shfl_xor(s, 8, 32);
      lrow[r] = lrow[r] * alpha[r] + s;
    }

    // P (D-layout) -> LDS -> reload in A-layout (wave-private bounce)
    __bf16* pl = &Pbuf[wave][0];
    #pragma unroll
    for (int r = 0; r < 8; ++r) {
      const int row = r + (hiL ? 8 : 0);
      pl[row * BJ + l16]      = (__bf16)P0[r];
      pl[row * BJ + 16 + l16] = (__bf16)P1[r];
    }
    lds_fence();
    V16U aP;
    {
      const int ko = hiL ? 16 : 0;
      aP.h[0] = *(const v8bf*)(pl + l16 * BJ + ko);
      aP.h[1] = *(const v8bf*)(pl + l16 * BJ + 8 + ko);
    }

    // O += P x V : 16 chunks of 16 channels, K=32 fully used
    #pragma unroll
    for (int ch = 0; ch < 16; ++ch) {
      #pragma unroll
      for (int r = 0; r < 8; ++r) acc[ch][r] *= alpha[r];
      const v16bf bV = *(const v16bf*)(vb + (ch * 16 + l16) * BJ + (hiL ? 16 : 0));
      acc[ch] = __builtin_amdgcn_wmma_f32_16x16x32_bf16(
          false, aP.v, false, bV, (short)0, acc[ch], false, false);
    }
    __syncthreads();
  }

  float inv[8];
  #pragma unroll
  for (int r = 0; r < 8; ++r) inv[r] = 1.f / lrow[r];
  #pragma unroll
  for (int ch = 0; ch < 16; ++ch) {
    const int cg = ch * 16 + l16;
    #pragma unroll
    for (int r = 0; r < 8; ++r) {
      const int ig = i0 + r + (hiL ? 8 : 0);
      out[((size_t)b * CCH + cg) * NTOK + ig] = acc[ch][r] * inv[r];
    }
  }
}

// ---------------- launcher --------------------------------------------------
extern "C" void kernel_launch(void* const* d_in, const int* in_sizes, int n_in,
                              void* d_out, int out_size, void* d_ws, size_t ws_size,
                              hipStream_t stream) {
  (void)in_sizes; (void)n_in; (void)out_size; (void)ws_size;
  const float* x  = (const float*)d_in[0];
  const float* y  = (const float*)d_in[1];
  const float* wq = (const float*)d_in[2];
  const float* bq = (const float*)d_in[3];
  const float* wk = (const float*)d_in[4];
  const float* bk = (const float*)d_in[5];
  const float* wv = (const float*)d_in[6];
  const float* bv = (const float*)d_in[7];
  float* out = (float*)d_out;

  char* ws = (char*)d_ws;
  size_t off = 0;
  auto walloc = [&](size_t bytes) {
    void* p = ws + off;
    off += (bytes + 255) & ~(size_t)255;
    return p;
  };
  __bf16* xb  = (__bf16*)walloc((size_t)NB * HH_ * WW_ * CCH * 2);  // 8 MB
  __bf16* yb  = (__bf16*)walloc((size_t)NB * HH_ * WW_ * CCH * 2);  // 8 MB
  __bf16* wqb = (__bf16*)walloc((size_t)9 * DQK * CCH * 2);
  __bf16* wkb = (__bf16*)walloc((size_t)9 * DQK * CCH * 2);
  __bf16* wvb = (__bf16*)walloc((size_t)9 * CCH * CCH * 2);          // 1.18 MB
  __bf16* Qp  = (__bf16*)walloc((size_t)NB * NTOK * DQK * 2);        // 512 KB
  __bf16* Kp  = (__bf16*)walloc((size_t)NB * NTOK * DQK * 2);        // 512 KB
  __bf16* Vt  = (__bf16*)walloc((size_t)NB * CCH * NTOK * 2);        // 8 MB

  // pack inputs to NHWC bf16
  pack_nhwc_kernel<<<(NB * CCH * HH_ * WW_) / 256, 256, 0, stream>>>(x, xb);
  pack_nhwc_kernel<<<(NB * CCH * HH_ * WW_) / 256, 256, 0, stream>>>(y, yb);
  // pack weights to [tap][OC][CI] bf16
  pack_w_kernel<<<(DQK * CCH * 9 + 255) / 256, 256, 0, stream>>>(wq, wqb, DQK);
  pack_w_kernel<<<(DQK * CCH * 9 + 255) / 256, 256, 0, stream>>>(wk, wkb, DQK);
  pack_w_kernel<<<(CCH * CCH * 9 + 255) / 256, 256, 0, stream>>>(wv, wvb, CCH);

  // convs: tiles = B*H*4*(OC/16); 8 wave-tiles per 256-thread block
  conv_wmma_kernel<<<(NB * HH_ * 4 * (DQK / 16)) / 8, 256, 0, stream>>>(xb, wqb, bq, Qp, DQK, 1);
  conv_wmma_kernel<<<(NB * HH_ * 4 * (DQK / 16)) / 8, 256, 0, stream>>>(yb, wkb, bk, Kp, DQK, 1);
  conv_wmma_kernel<<<(NB * HH_ * 4 * (CCH / 16)) / 8, 256, 0, stream>>>(yb, wvb, bv, Vt, CCH, 0);

  // fused flash attention
  attn_kernel<<<NB * (NTOK / 64), 128, 0, stream>>>(Qp, Kp, Vt, out);
}
